// RelationshipEmbedding_38706245272356
// MI455X (gfx1250) — matrix operand order repaired and checked by reading, not compiled, not run on hardware
//
#include <hip/hip_runtime.h>
#include <hip/hip_bf16.h>

// ---------------------------------------------------------------------------
// RelationshipEmbedding on gfx1250 via v_wmma_f32_16x16x32_f16.
// Algebraic folds: emb-part of layer1 -> constant vector; concat([x,x])@W1 ->
// x@(W1a+W1b); fs@W_rs.T@Wd.T -> fs@(Wd@W_rs).T; pred bias precomputed.
// Dominant 262144x512x256 GEMM done per-(t,i) with h staged in 64KB LDS.
// Round 2: software-pipelined fragment loads so WMMAs overlap VMEM latency
// (batch B-frag clauses + prefetch next chunk) instead of s_wait_loadcnt 0
// before every WMMA.  (i3d_feats input is unused by the reference.)
// ---------------------------------------------------------------------------

typedef _Float16 half_t;
typedef __attribute__((ext_vector_type(8)))  _Float16 v8h;
typedef __attribute__((ext_vector_type(16))) _Float16 v16h;
typedef __attribute__((ext_vector_type(8)))  float    v8f;

#define N_TOK   4096      // T*M
#define K_IN    2080      // 2048 roi + 4 spatial + 28 zero pad (mult of 32)
#define H_DIM   512
#define D_DIM   256

// ---- fragment loaders (layouts per CDNA5 ISA 7.12.2) ----------------------
// f16 A 16x32: lanes 0-15 hold row M=lane, K = {0..7, 16..23};
//              lanes 16-31 hold row M=lane-16, K = {8..15, 24..31}.
__device__ inline v16h ld_afrag(const half_t* __restrict__ row, int k0, int lane) {
  const int off = (lane < 16) ? 0 : 8;
  v8h lo = *(const v8h*)(row + k0 + off);
  v8h hi = *(const v8h*)(row + k0 + 16 + off);
  v16h a;
#pragma unroll
  for (int i = 0; i < 8; ++i) { a[i] = lo[i]; a[i + 8] = hi[i]; }
  return a;
}

// f16 B 32x16: lane holds col N=lane%16; lanes 0-15 K={0..7,16..23}, lanes 16-31
// K={8..15,24..31} (same K order as the A fragment).
__device__ inline v16h ld_bfrag(const half_t* __restrict__ wrow, int k0, int lane) {
  const int koff = (lane < 16) ? 0 : 8;
  v8h lo = *(const v8h*)(wrow + k0 + koff);
  v8h hi = *(const v8h*)(wrow + k0 + koff + 16);
  v16h b;
#pragma unroll
  for (int i = 0; i < 8; ++i) { b[i] = lo[i]; b[i + 8] = hi[i]; }
  return b;
}

// ---- generic WMMA GEMM: out[r,c] = act( A[r,:]·W[c,:] + addv[c] ) ---------
// A: Nr x K f16 row-major.  W: Nc x K f16 row-major (i.e. x @ W.T).
// Block = 256 thr (8 waves). WG tile = 128 rows x 64 cols; wave = 16 x 64.
// K-loop is software-pipelined: frags for chunk k+1 prefetched during the
// WMMAs of chunk k.
__global__ __launch_bounds__(256) void gemm_wmma_f16(
    const half_t* __restrict__ A, const half_t* __restrict__ W,
    const float* __restrict__ addv,
    float* __restrict__ outF32, half_t* __restrict__ outF16,
    int Nc, int K, int relu)
{
  const int lane  = threadIdx.x & 31;
  const int wv    = threadIdx.x >> 5;
  const int rbase = blockIdx.x * 128 + wv * 16;
  const int nbase = blockIdx.y * 64;
  const int m     = lane & 15;

  v8f acc[4] = {};
  const half_t* arow = A + (size_t)(rbase + m) * K;
  const half_t* wr[4];
#pragma unroll
  for (int tt = 0; tt < 4; ++tt)
    wr[tt] = W + (size_t)(nbase + tt * 16 + m) * K;

  const int nchunks = K >> 5;

  // prologue: chunk 0
  v16h a_cur = ld_afrag(arow, 0, lane);
  v16h b_cur[4];
#pragma unroll
  for (int tt = 0; tt < 4; ++tt) b_cur[tt] = ld_bfrag(wr[tt], 0, lane);

#pragma unroll 2
  for (int c = 0; c < nchunks - 1; ++c) {
    const int kn = (c + 1) << 5;
    // prefetch chunk c+1
    v16h a_nxt = ld_afrag(arow, kn, lane);
    v16h b_nxt[4];
#pragma unroll
    for (int tt = 0; tt < 4; ++tt) b_nxt[tt] = ld_bfrag(wr[tt], kn, lane);
    // compute chunk c
#pragma unroll
    for (int tt = 0; tt < 4; ++tt)
      acc[tt] = __builtin_amdgcn_wmma_f32_16x16x32_f16(
          false, a_cur, false, b_nxt ? b_cur[tt] : b_cur[tt], (short)0, acc[tt], false, false);
    a_cur = a_nxt;
#pragma unroll
    for (int tt = 0; tt < 4; ++tt) b_cur[tt] = b_nxt[tt];
  }
  // epilogue chunk
#pragma unroll
  for (int tt = 0; tt < 4; ++tt)
    acc[tt] = __builtin_amdgcn_wmma_f32_16x16x32_f16(
        false, a_cur, false, b_cur[tt], (short)0, acc[tt], false, false);

  const int rowoff = (lane < 16) ? 0 : 8;
#pragma unroll
  for (int tt = 0; tt < 4; ++tt) {
    const int col = nbase + tt * 16 + m;
    const float av = addv ? addv[col] : 0.0f;
#pragma unroll
    for (int r = 0; r < 8; ++r) {
      const size_t row = (size_t)(rbase + r + rowoff);
      float v = acc[tt][r] + av;
      if (relu) v = fmaxf(v, 0.0f);
      if (outF16) outF16[row * Nc + col] = (half_t)v;
      else        outF32[row * Nc + col] = v;
    }
  }
}

// ---- dominant relation kernel: one WG per (t,i) ---------------------------
// h[j,k] = relu(A[t*64+i,k] - B[t*64+j,k]) staged f16 in LDS (64 KB),
// out[(t,i,j),n] = h[j,:]·W2[n,:] + b2[n].  8 waves: wave = 16 rows x 128 cols.
// Per chunk: one clause of 16 global b128 loads (8 B-frags), next A-frag
// prefetched from LDS, then 8 WMMAs consume in order (partial waits).
__global__ __launch_bounds__(256) void rel_wmma(
    const float* __restrict__ Abuf, const float* __restrict__ Bbuf,
    const half_t* __restrict__ W2, const float* __restrict__ b2,
    float* __restrict__ out)
{
  __shared__ half_t hsh[64 * 512];
  const int ti = blockIdx.x;        // t*64 + i
  const int t  = ti >> 6;
  const float* Arow = Abuf + (size_t)ti * 512;
  const float* Bfrm = Bbuf + (size_t)(t * 64) * 512;

  for (int idx = threadIdx.x; idx < 64 * 512; idx += 256) {
    const int j = idx >> 9, k = idx & 511;
    const float v = Arow[k] - Bfrm[(size_t)j * 512 + k];
    hsh[idx] = (half_t)fmaxf(v, 0.0f);
  }
  __syncthreads();

  const int lane = threadIdx.x & 31;
  const int wv   = threadIdx.x >> 5;
  const int jb   = (wv & 3) * 16;       // row (j) block
  const int nb   = (wv >> 2) * 128;     // col half
  const int m    = lane & 15;

  v8f acc[8] = {};
  const half_t* hrow = hsh + (size_t)(jb + m) * 512;
  const half_t* wr[8];
#pragma unroll
  for (int tt = 0; tt < 8; ++tt)
    wr[tt] = W2 + (size_t)(nb + tt * 16 + m) * 512;

  v16h a_cur = ld_afrag(hrow, 0, lane);
  for (int c = 0; c < 15; ++c) {
    v16h b[8];
#pragma unroll
    for (int tt = 0; tt < 8; ++tt) b[tt] = ld_bfrag(wr[tt], c << 5, lane);
    v16h a_nxt = ld_afrag(hrow, (c + 1) << 5, lane);
#pragma unroll
    for (int tt = 0; tt < 8; ++tt)
      acc[tt] = __builtin_amdgcn_wmma_f32_16x16x32_f16(
          false, a_cur, false, b[tt], (short)0, acc[tt], false, false);
    a_cur = a_nxt;
  }
  {
    v16h b[8];
#pragma unroll
    for (int tt = 0; tt < 8; ++tt) b[tt] = ld_bfrag(wr[tt], 15 << 5, lane);
#pragma unroll
    for (int tt = 0; tt < 8; ++tt)
      acc[tt] = __builtin_amdgcn_wmma_f32_16x16x32_f16(
          false, a_cur, false, b[tt], (short)0, acc[tt], false, false);
  }

  const int rowoff = (lane < 16) ? 0 : 8;
  const size_t obase = (size_t)ti * 64;   // output row base = (t*64+i)*64
#pragma unroll
  for (int tt = 0; tt < 8; ++tt) {
    const int col = nb + tt * 16 + m;
    const float bias = b2[col];
#pragma unroll
    for (int r = 0; r < 8; ++r) {
      const int j = jb + r + rowoff;
      out[(obase + j) * 256 + col] = acc[tt][r] + bias;
    }
  }
}

// ---- small prep kernels ---------------------------------------------------
__global__ void pack_x(const float* __restrict__ roi, const float* __restrict__ spa,
                       half_t* __restrict__ X) {
  size_t idx = (size_t)blockIdx.x * 256 + threadIdx.x;
  if (idx >= (size_t)N_TOK * K_IN) return;
  int n = (int)(idx / K_IN), c = (int)(idx % K_IN);
  float v = (c < 2048) ? roi[(size_t)n * 2048 + c]
          : (c < 2052) ? spa[(size_t)n * 4 + (c - 2048)] : 0.0f;
  X[idx] = (half_t)v;
}

__global__ void pack_w1(const float* __restrict__ w1, half_t* __restrict__ Wh) {
  size_t idx = (size_t)blockIdx.x * 256 + threadIdx.x;
  if (idx >= (size_t)H_DIM * K_IN) return;
  int h = (int)(idx / K_IN), c = (int)(idx % K_IN);
  Wh[idx] = (half_t)((c < 2052) ? w1[(size_t)h * 2352 + c] : 0.0f);
}

__global__ void cvt_f16(const float* __restrict__ src, half_t* __restrict__ dst, int n) {
  int idx = blockIdx.x * 256 + threadIdx.x;
  if (idx < n) dst[idx] = (half_t)src[idx];
}

// fold concat([x,x]) weight: out[h,d] = w1[h,d] + w1[h,256+d]   (512x256)
__global__ void fold_fuse(const float* __restrict__ fw1, half_t* __restrict__ Wh) {
  int idx = blockIdx.x * 256 + threadIdx.x;
  if (idx >= H_DIM * D_DIM) return;
  int h = idx >> 8, d = idx & 255;
  Wh[idx] = (half_t)(fw1[(size_t)h * 512 + d] + fw1[(size_t)h * 512 + 256 + d]);
}

// outv[h] = b[h] + sum_k W[h, off+k] * emb[k]
__global__ void headvec(const float* __restrict__ W, int rowstride, int off, int len,
                        const float* __restrict__ emb, const float* __restrict__ b,
                        float* __restrict__ outv) {
  int h = blockIdx.x * 256 + threadIdx.x;
  if (h >= H_DIM) return;
  float s = b[h];
  const float* row = W + (size_t)h * rowstride + off;
  for (int k = 0; k < len; ++k) s += row[k] * emb[k];
  outv[h] = s;
}

// M[a,e] = sum_d rel_w1[a,d] * Wr[d,e]   (Wd @ W_r), a<512, e<256, d<256
__global__ void ms_kern(const float* __restrict__ relw1, const float* __restrict__ Wr,
                        half_t* __restrict__ M) {
  int idx = blockIdx.x * 256 + threadIdx.x;
  if (idx >= H_DIM * D_DIM) return;
  int a = idx >> 8, e = idx & 255;
  float s = 0.0f;
  const float* wdrow = relw1 + (size_t)a * 556;
  for (int d = 0; d < 256; ++d) s += wdrow[d] * Wr[(size_t)d * 256 + e];
  M[idx] = (half_t)s;
}

// ---------------------------------------------------------------------------
extern "C" void kernel_launch(void* const* d_in, const int* in_sizes, int n_in,
                              void* d_out, int out_size, void* d_ws, size_t ws_size,
                              hipStream_t stream) {
  const float* roi     = (const float*)d_in[0];
  const float* spatial = (const float*)d_in[1];
  // d_in[2] = i3d_feats: unused by reference
  const float* subj_emb = (const float*)d_in[3];
  const float* obj_emb  = (const float*)d_in[4];
  const float* pred_emb = (const float*)d_in[5];
  const float* subj_w1 = (const float*)d_in[6];
  const float* subj_b1 = (const float*)d_in[7];
  const float* subj_w2 = (const float*)d_in[8];
  const float* subj_b2 = (const float*)d_in[9];
  const float* obj_w1  = (const float*)d_in[10];
  const float* obj_b1  = (const float*)d_in[11];
  const float* obj_w2  = (const float*)d_in[12];
  const float* obj_b2  = (const float*)d_in[13];
  const float* fuse_s_w1 = (const float*)d_in[14];
  const float* fuse_s_b1 = (const float*)d_in[15];
  const float* fuse_s_w2 = (const float*)d_in[16];
  const float* fuse_s_b2 = (const float*)d_in[17];
  const float* fuse_o_w1 = (const float*)d_in[18];
  const float* fuse_o_b1 = (const float*)d_in[19];
  const float* fuse_o_w2 = (const float*)d_in[20];
  const float* fuse_o_b2 = (const float*)d_in[21];
  const float* W_rs   = (const float*)d_in[22];
  const float* W_ro   = (const float*)d_in[23];
  const float* rel_w1 = (const float*)d_in[24];
  const float* rel_b1 = (const float*)d_in[25];
  const float* rel_w2 = (const float*)d_in[26];
  const float* rel_b2 = (const float*)d_in[27];
  float* out = (float*)d_out;

  // ---- workspace carve-up (all 256B aligned) ----
  char* p = (char*)d_ws;
  auto alloc = [&](size_t bytes) { char* q = p; p += (bytes + 255) & ~(size_t)255; return q; };
  half_t* Xh    = (half_t*)alloc((size_t)N_TOK * K_IN * 2);
  half_t* W1sh  = (half_t*)alloc((size_t)H_DIM * K_IN * 2);
  half_t* W1oh  = (half_t*)alloc((size_t)H_DIM * K_IN * 2);
  half_t* W2sh  = (half_t*)alloc((size_t)D_DIM * H_DIM * 2);
  half_t* W2oh  = (half_t*)alloc((size_t)D_DIM * H_DIM * 2);
  half_t* Wfs1h = (half_t*)alloc((size_t)H_DIM * D_DIM * 2);
  half_t* Wfo1h = (half_t*)alloc((size_t)H_DIM * D_DIM * 2);
  half_t* Wfs2h = (half_t*)alloc((size_t)D_DIM * H_DIM * 2);
  half_t* Wfo2h = (half_t*)alloc((size_t)D_DIM * H_DIM * 2);
  half_t* Msh   = (half_t*)alloc((size_t)H_DIM * D_DIM * 2);
  half_t* Moh   = (half_t*)alloc((size_t)H_DIM * D_DIM * 2);
  half_t* W2rel = (half_t*)alloc((size_t)D_DIM * H_DIM * 2);
  float*  c_subj = (float*)alloc(H_DIM * 4);
  float*  c_obj  = (float*)alloc(H_DIM * 4);
  float*  bias_h = (float*)alloc(H_DIM * 4);
  half_t* t1s  = (half_t*)alloc((size_t)N_TOK * H_DIM * 2);  // layer1 out / reused as u
  half_t* t1o  = (half_t*)alloc((size_t)N_TOK * H_DIM * 2);
  half_t* fsb  = (half_t*)alloc((size_t)N_TOK * D_DIM * 2);  // fs_base / reused as fs
  half_t* fob  = (half_t*)alloc((size_t)N_TOK * D_DIM * 2);
  float*  Abuf = (float*)alloc((size_t)N_TOK * H_DIM * 4);
  float*  Bbuf = (float*)alloc((size_t)N_TOK * H_DIM * 4);

  // ---- prep ----
  {
    int nb = (int)(((size_t)N_TOK * K_IN + 255) / 256);
    pack_x<<<nb, 256, 0, stream>>>(roi, spatial, Xh);
  }
  {
    int nb = (int)(((size_t)H_DIM * K_IN + 255) / 256);
    pack_w1<<<nb, 256, 0, stream>>>(subj_w1, W1sh);
    pack_w1<<<nb, 256, 0, stream>>>(obj_w1, W1oh);
  }
  {
    int n = D_DIM * H_DIM, nb = (n + 255) / 256;
    cvt_f16<<<nb, 256, 0, stream>>>(subj_w2, W2sh, n);
    cvt_f16<<<nb, 256, 0, stream>>>(obj_w2, W2oh, n);
    cvt_f16<<<nb, 256, 0, stream>>>(fuse_s_w2, Wfs2h, n);
    cvt_f16<<<nb, 256, 0, stream>>>(fuse_o_w2, Wfo2h, n);
    cvt_f16<<<nb, 256, 0, stream>>>(rel_w2, W2rel, n);
  }
  {
    int nb = (H_DIM * D_DIM + 255) / 256;
    fold_fuse<<<nb, 256, 0, stream>>>(fuse_s_w1, Wfs1h);
    fold_fuse<<<nb, 256, 0, stream>>>(fuse_o_w1, Wfo1h);
    ms_kern<<<nb, 256, 0, stream>>>(rel_w1, W_rs, Msh);
    ms_kern<<<nb, 256, 0, stream>>>(rel_w1, W_ro, Moh);
  }
  headvec<<<2, 256, 0, stream>>>(subj_w1, 2352, 2052, 300, subj_emb, subj_b1, c_subj);
  headvec<<<2, 256, 0, stream>>>(obj_w1, 2352, 2052, 300, obj_emb, obj_b1, c_obj);
  headvec<<<2, 256, 0, stream>>>(rel_w1, 556, 256, 300, pred_emb, rel_b1, bias_h);

  // ---- MLP chain (all WMMA) ----
  dim3 blk(256);
  // layer1: relu(X @ W1.T + c)      4096 x 2080 -> 512
  gemm_wmma_f16<<<dim3(32, 8), blk, 0, stream>>>(Xh, W1sh, c_subj, nullptr, t1s, H_DIM, K_IN, 1);
  gemm_wmma_f16<<<dim3(32, 8), blk, 0, stream>>>(Xh, W1oh, c_obj, nullptr, t1o, H_DIM, K_IN, 1);
  // layer2: t1 @ w2.T + b2          4096 x 512 -> 256
  gemm_wmma_f16<<<dim3(32, 4), blk, 0, stream>>>(t1s, W2sh, subj_b2, nullptr, fsb, D_DIM, H_DIM, 0);
  gemm_wmma_f16<<<dim3(32, 4), blk, 0, stream>>>(t1o, W2oh, obj_b2, nullptr, fob, D_DIM, H_DIM, 0);
  // fuse layer1 (folded): relu(fb @ (W1a+W1b).T + b1)   4096 x 256 -> 512  (into t1*)
  gemm_wmma_f16<<<dim3(32, 8), blk, 0, stream>>>(fsb, Wfs1h, fuse_s_b1, nullptr, t1s, H_DIM, D_DIM, 1);
  gemm_wmma_f16<<<dim3(32, 8), blk, 0, stream>>>(fob, Wfo1h, fuse_o_b1, nullptr, t1o, H_DIM, D_DIM, 1);
  // fuse layer2: u @ w2.T + b2      4096 x 512 -> 256  (into fsb/fob)
  gemm_wmma_f16<<<dim3(32, 4), blk, 0, stream>>>(t1s, Wfs2h, fuse_s_b2, nullptr, fsb, D_DIM, H_DIM, 0);
  gemm_wmma_f16<<<dim3(32, 4), blk, 0, stream>>>(t1o, Wfo2h, fuse_o_b2, nullptr, fob, D_DIM, H_DIM, 0);
  // projection fold: A = fs @ (Wd@W_rs).T + bias_h ; B = fo @ (Wd@W_ro).T   -> f32
  gemm_wmma_f16<<<dim3(32, 8), blk, 0, stream>>>(fsb, Msh, bias_h, Abuf, nullptr, H_DIM, D_DIM, 0);
  gemm_wmma_f16<<<dim3(32, 8), blk, 0, stream>>>(fob, Moh, nullptr, Bbuf, nullptr, H_DIM, D_DIM, 0);

  // ---- dominant relation GEMM: 4096 WGs, 68.7 GFLOP ----
  rel_wmma<<<dim3(4096), blk, 0, stream>>>(Abuf, Bbuf, W2rel, rel_b2, out);

  (void)in_sizes; (void)n_in; (void)out_size; (void)ws_size;
}
